// TP_lstm_32950989095134
// MI455X (gfx1250) — matrix operand order repaired and checked by reading, not compiled, not run on hardware
//
#include <hip/hip_runtime.h>
#include <hip/hip_bf16.h>

// CDNA5 / gfx1250: wave32, WMMA 16x16x32 f16 -> f32 accum.

typedef __attribute__((ext_vector_type(16))) _Float16 v16h;
typedef __attribute__((ext_vector_type(8)))  _Float16 v8h;
typedef __attribute__((ext_vector_type(8)))  float    v8f;

#define WMMA16(a, b, c) \
    __builtin_amdgcn_wmma_f32_16x16x32_f16(false, (a), false, (b), (short)0, (c), false, false)

// Fast activations: v_exp_f32 + v_rcp_f32 (no IEEE div sequence).
__device__ __forceinline__ float sigm(float x) {
    return __builtin_amdgcn_rcpf(1.0f + __expf(-x));
}
__device__ __forceinline__ float tanh_fast(float x) {
    return 2.0f * __builtin_amdgcn_rcpf(1.0f + __expf(-2.0f * x)) - 1.0f;
}

// A-fragment (16x32 f16): lane l<16 holds row M=l, K = kb+{0..7, 16..23};
// lane l>=16 holds row M=l-16, K = kb+{8..15, 24..31}. Two 16B LDS loads.
__device__ __forceinline__ v16h ldsA(const _Float16* base, int row, int kb, int lane) {
    int koff = (lane >> 4) << 3;                 // 0 or 8
    const _Float16* p = base + row * 64 + kb + koff;
    v8h lo = *(const v8h*)(p);
    v8h hi = *(const v8h*)(p + 16);
    return __builtin_shufflevector(lo, hi, 0,1,2,3,4,5,6,7,8,9,10,11,12,13,14,15);
}

// B-fragment (32x16 f16): lane l<16 holds col N=l, K = kb+{0..7, 16..23};
// lane l>=16 holds col N=l-16, K = kb+{8..15, 24..31}. W stored [N][K] row-major,
// so each lane reads two 8-half chunks of W's row.
__device__ __forceinline__ v16h ldsB(const _Float16* W, int K, int ncol, int kb, int lane) {
    int k0 = kb + ((lane >> 4) << 3);            // +0 or +8
    const _Float16* p = W + ncol * K + k0;
    v8h lo = *(const v8h*)(p);
    v8h hi = *(const v8h*)(p + 16);
    return __builtin_shufflevector(lo, hi, 0,1,2,3,4,5,6,7,8,9,10,11,12,13,14,15);
}

#define NB 128
#define NS 128
#define NT 64
#define NH 64
#define NOH 32
#define TPRED 8

// One 16x16 spatial-gate tile: WHICH=0 -> f_s (scales c), WHICH=1 -> i_s (builds h_g).
// WHICH is a template arg so the writeback path is branch-free at runtime.
template <int WHICH>
__device__ __forceinline__ void gemm1_tile(const _Float16* __restrict__ sh_h,
                                           float* __restrict__ sh_c,
                                           _Float16* __restrict__ sh_hg,
                                           const _Float16* __restrict__ W,
                                           const float* __restrict__ bias,
                                           int mt, int nt, int lane, int lhalf, int hi16) {
    const int ncol = nt * 16 + lhalf;
    const float bv = bias[ncol];
    v8f acc = { bv, bv, bv, bv, bv, bv, bv, bv };
    #pragma unroll
    for (int kt = 0; kt < 4; ++kt) {
        // k<64 -> h_after = padded row s+2; k>=64 -> h_before = padded row s
        const int srow = mt * 16 + lhalf + (kt < 2 ? 2 : 0);
        const int kbA  = (kt < 2) ? kt * 32 : (kt - 2) * 32;
        v16h a  = ldsA(sh_h, srow, kbA, lane);
        v16h bw = ldsB(W, 128, ncol, kt * 32, lane);
        acc = WMMA16(a, bw, acc);
    }
    #pragma unroll
    for (int r = 0; r < 8; ++r) {
        const int s   = mt * 16 + r + hi16 * 8;
        const int off = s * 64 + ncol;
        const float g = sigm(acc[r]);
        if (WHICH == 0) {
            sh_c[off] *= g;                                           // c_g = c * f_s
        } else {
            sh_hg[off] = (_Float16)((float)sh_h[(s + 1) * 64 + ncol] * g); // h_g = h * i_s
        }
    }
}

__global__ __launch_bounds__(512, 1)
void tp_lstm_kernel(const float* __restrict__ xin,   // [B,S,T,3]
                    const float* __restrict__ Wih,   // [256,3]
                    const float* __restrict__ bih,   // [256]
                    const float* __restrict__ Whh,   // [256,64]
                    const float* __restrict__ bhh,   // [256]
                    const float* __restrict__ Wsf,   // [64,128]
                    const float* __restrict__ bsf,   // [64]
                    const float* __restrict__ Wsi,   // [64,128]
                    const float* __restrict__ bsi,   // [64]
                    const float* __restrict__ Wo1,   // [32,64]
                    const float* __restrict__ bo1,   // [32]
                    const float* __restrict__ Wo2,   // [1,32]
                    const float* __restrict__ bo2,   // [1]
                    float* __restrict__ out)         // [B,S,56]
{
    // LDS state: h padded with one zero row at each end so the +/-1 spatial
    // shift is a pure row offset (h[s] lives at padded row s+1).
    __shared__ _Float16 sh_h[(NS + 2) * NH];     // 16.6 KB  f16 hidden state
    __shared__ float    sh_c[NS * NH];           // 32   KB  f32 cell state
    __shared__ _Float16 sh_hg[NS * NH];          // 16   KB  f16 gated hidden
    __shared__ _Float16 sh_Wsf[64 * 128];        // 16   KB
    __shared__ _Float16 sh_Wsi[64 * 128];        // 16   KB
    __shared__ _Float16 sh_Whh[256 * 64];        // 32   KB
    __shared__ _Float16 sh_Wo1[32 * 64];         // 4    KB
    __shared__ float    sh_Wih[256 * 3];
    __shared__ float    sh_x[NS * 3];            // staged x slice for current t
    __shared__ float    sh_bsf[64], sh_bsi[64], sh_bihh[256], sh_bo1[32], sh_wo2[32];
    __shared__ float    sh_oacc[NS];
    __shared__ float    sh_bo2;

    const int tid   = threadIdx.x;
    const int wv    = tid >> 5;         // wave id 0..15
    const int lane  = tid & 31;
    const int lhalf = lane & 15;
    const int hi16  = lane >> 4;        // 0 or 1
    const int bb    = blockIdx.x;       // batch row

    // ---- one-time: stage weights (f32->f16), biases, zero state ----
    for (int i = tid; i < 64 * 128; i += 512) {
        sh_Wsf[i] = (_Float16)Wsf[i];
        sh_Wsi[i] = (_Float16)Wsi[i];
    }
    for (int i = tid; i < 256 * 64; i += 512) sh_Whh[i] = (_Float16)Whh[i];
    for (int i = tid; i < 32 * 64;  i += 512) sh_Wo1[i] = (_Float16)Wo1[i];
    for (int i = tid; i < 256 * 3;  i += 512) sh_Wih[i] = Wih[i];
    if (tid < 256) sh_bihh[tid] = bih[tid] + bhh[tid];
    if (tid < 64)  { sh_bsf[tid] = bsf[tid]; sh_bsi[tid] = bsi[tid]; }
    if (tid < 32)  { sh_bo1[tid] = bo1[tid]; sh_wo2[tid] = Wo2[tid]; }
    if (tid == 0)  sh_bo2 = bo2[0];
    for (int i = tid; i < (NS + 2) * NH; i += 512) sh_h[i] = (_Float16)0.0f;
    for (int i = tid; i < NS * NH;       i += 512) sh_c[i] = 0.0f;
    __syncthreads();

    for (int t = 0; t < NT; ++t) {
        // Stage x[:, t, :] into LDS; the global loads overlap phase A and are
        // ordered before phase B by the B1 barrier.
        if (tid < NS * 3) {
            const int s = tid / 3, comp = tid - s * 3;
            sh_x[tid] = xin[((size_t)(bb * NS + s) * NT + t) * 3 + comp];
        }

        // ================= Phase A: spatial gates =================
        // f_s = sigmoid(sg @ Wsf^T + bsf), i_s likewise. sg[s] = [h[s+1] | h[s-1]].
        // 64 tiles: which(2) x mt(8) x nt(4); wave wv owns rem = wv, wv+16 per which.
        #pragma unroll
        for (int rep = 0; rep < 2; ++rep) {
            const int rem = wv + rep * 16;       // 0..31
            gemm1_tile<0>(sh_h, sh_c, sh_hg, sh_Wsf, sh_bsf,
                          rem >> 2, rem & 3, lane, lhalf, hi16);
        }
        #pragma unroll
        for (int rep = 0; rep < 2; ++rep) {
            const int rem = wv + rep * 16;
            gemm1_tile<1>(sh_h, sh_c, sh_hg, sh_Wsi, sh_bsi,
                          rem >> 2, rem & 3, lane, lhalf, hi16);
        }
        __syncthreads();   // B1

        // ================= Phase B: LSTM gates + cell update =================
        // gates = x@Wih^T + h_g@Whh^T + (bih+bhh); one wave owns i/f/g/o tiles
        // of a 16x16 (s,h) block so the whole nonlinearity stays in registers.
        for (int p = wv; p < 32; p += 16) {
            const int mt = p >> 2;
            const int j  = p & 3;
            const int colI = 0   + j * 16 + lhalf;
            const int colF = 64  + j * 16 + lhalf;
            const int colG = 128 + j * 16 + lhalf;
            const int colO = 192 + j * 16 + lhalf;
            v8f aI, aF, aG, aO;
            #pragma unroll
            for (int r = 0; r < 8; ++r) {
                const int s = mt * 16 + r + hi16 * 8;
                const float x0 = sh_x[s * 3], x1 = sh_x[s * 3 + 1], x2 = sh_x[s * 3 + 2];
                aI[r] = sh_bihh[colI] + sh_Wih[colI*3]*x0 + sh_Wih[colI*3+1]*x1 + sh_Wih[colI*3+2]*x2;
                aF[r] = sh_bihh[colF] + sh_Wih[colF*3]*x0 + sh_Wih[colF*3+1]*x1 + sh_Wih[colF*3+2]*x2;
                aG[r] = sh_bihh[colG] + sh_Wih[colG*3]*x0 + sh_Wih[colG*3+1]*x1 + sh_Wih[colG*3+2]*x2;
                aO[r] = sh_bihh[colO] + sh_Wih[colO*3]*x0 + sh_Wih[colO*3+1]*x1 + sh_Wih[colO*3+2]*x2;
            }
            #pragma unroll
            for (int kt = 0; kt < 2; ++kt) {
                v16h a = ldsA(sh_hg, mt * 16 + lhalf, kt * 32, lane);
                aI = WMMA16(a, ldsB(sh_Whh, 64, colI, kt * 32, lane), aI);
                aF = WMMA16(a, ldsB(sh_Whh, 64, colF, kt * 32, lane), aF);
                aG = WMMA16(a, ldsB(sh_Whh, 64, colG, kt * 32, lane), aG);
                aO = WMMA16(a, ldsB(sh_Whh, 64, colO, kt * 32, lane), aO);
            }
            #pragma unroll
            for (int r = 0; r < 8; ++r) {
                const int s  = mt * 16 + r + hi16 * 8;
                const int hc = j * 16 + lhalf;
                const float ig = sigm(aI[r]);
                const float fg = sigm(aF[r]);
                const float gg = tanh_fast(aG[r]);
                const float og = sigm(aO[r]);
                const float cn = fg * sh_c[s * 64 + hc] + ig * gg;   // sh_c holds c_g
                sh_c[s * 64 + hc] = cn;
                sh_h[(s + 1) * 64 + hc] = (_Float16)(og * tanh_fast(cn));
            }
        }
        __syncthreads();   // B2
        if (tid < NS) sh_oacc[tid] = sh_bo2;
        __syncthreads();   // B3

        // ================= Phase D: output head =================
        // hid = relu(h_new @ Wo1^T + bo1); out[s] = hid . wo2 + bo2.
        // 16 tiles (mt 0..7, nt 0..1), one per wave.
        {
            const int mt = wv >> 1;
            const int nt = wv & 1;
            const int ncol = nt * 16 + lhalf;
            const float bv = sh_bo1[ncol];
            v8f acc = { bv, bv, bv, bv, bv, bv, bv, bv };
            #pragma unroll
            for (int kt = 0; kt < 2; ++kt) {
                v16h a  = ldsA(sh_h, mt * 16 + lhalf + 1, kt * 32, lane); // h_new, unshifted
                v16h bw = ldsB(sh_Wo1, 64, ncol, kt * 32, lane);
                acc = WMMA16(a, bw, acc);
            }
            const float w2 = sh_wo2[ncol];
            #pragma unroll
            for (int r = 0; r < 8; ++r) {
                float pv = (acc[r] > 0.0f ? acc[r] : 0.0f) * w2;
                pv += __shfl_xor(pv, 1);
                pv += __shfl_xor(pv, 2);
                pv += __shfl_xor(pv, 4);
                pv += __shfl_xor(pv, 8);    // sum over the 16-lane half (16 output cols)
                if (lhalf == 0)
                    atomicAdd(&sh_oacc[mt * 16 + r + hi16 * 8], pv);
            }
        }
        __syncthreads();   // B4

        if (t >= TPRED && tid < NS) {
            out[((size_t)bb * NS + tid) * (NT - TPRED) + (t - TPRED)] = sh_oacc[tid];
        }
    }
}

extern "C" void kernel_launch(void* const* d_in, const int* in_sizes, int n_in,
                              void* d_out, int out_size, void* d_ws, size_t ws_size,
                              hipStream_t stream) {
    (void)in_sizes; (void)n_in; (void)d_ws; (void)ws_size; (void)out_size;
    const float* xin = (const float*)d_in[0];
    // d_in[1] = lane (unused by the reference computation)
    const float* Wih = (const float*)d_in[2];
    const float* bih = (const float*)d_in[3];
    const float* Whh = (const float*)d_in[4];
    const float* bhh = (const float*)d_in[5];
    const float* Wsf = (const float*)d_in[6];
    const float* bsf = (const float*)d_in[7];
    const float* Wsi = (const float*)d_in[8];
    const float* bsi = (const float*)d_in[9];
    const float* Wo1 = (const float*)d_in[10];
    const float* bo1 = (const float*)d_in[11];
    const float* Wo2 = (const float*)d_in[12];
    const float* bo2 = (const float*)d_in[13];
    float* out = (float*)d_out;

    hipLaunchKernelGGL(tp_lstm_kernel, dim3(NB), dim3(512), 0, stream,
                       xin, Wih, bih, Whh, bhh, Wsf, bsf, Wsi, bsi,
                       Wo1, bo1, Wo2, bo2, out);
}
